// Discriminator_31430570672344
// MI455X (gfx1250) — matrix operand order repaired and checked by reading, not compiled, not run on hardware
//
#include <hip/hip_runtime.h>
#include <hip/hip_bf16.h>
#include <stdint.h>

// ---------------------------------------------------------------------------
// Types for CDNA5 WMMA (wave32, 16x16x32 bf16 -> f32)
// ---------------------------------------------------------------------------
typedef __bf16 bf16;
typedef bf16  v16bf __attribute__((ext_vector_type(16)));
typedef float v8f   __attribute__((ext_vector_type(8)));

#define GN_NODES 102400
#define GN_R     3
#define GN_E     819200

// ---------------------------------------------------------------------------
// Utility kernels
// ---------------------------------------------------------------------------
__global__ void fill_f32(float* __restrict__ p, float v, int n) {
    int i = blockIdx.x * blockDim.x + threadIdx.x;
    if (i < n) p[i] = v;
}

// deg accumulation for one big launch over R*E edges
__global__ void degree_kernel(const int* __restrict__ src, const int* __restrict__ dst,
                              float* __restrict__ degOut, float* __restrict__ degIn) {
    int i = blockIdx.x * blockDim.x + threadIdx.x;
    if (i >= GN_R * GN_E) return;
    int r = i / GN_E;
    atomicAdd(&degOut[(size_t)r * GN_NODES + src[i]], 1.0f);
    atomicAdd(&degIn [(size_t)r * GN_NODES + dst[i]], 1.0f);
}

__global__ void rsqrt_clip_kernel(float* __restrict__ p, int n) {
    int i = blockIdx.x * blockDim.x + threadIdx.x;
    if (i < n) {
        float d = p[i];
        p[i] = rsqrtf(d < 1.0f ? 1.0f : d);
    }
}

// C[n*OUT + j] = sum_r bias[r*OUT + j]
__global__ void bias_init_kernel(float* __restrict__ C, const float* __restrict__ bias,
                                 int OUT, int total) {
    int i = blockIdx.x * blockDim.x + threadIdx.x;
    if (i >= total) return;
    int j = i & (OUT - 1);
    float s = 0.f;
#pragma unroll
    for (int r = 0; r < GN_R; ++r) s += bias[r * OUT + j];
    C[i] = s;
}

// Wt[r][o][k] = (bf16) W[r][k][o]    (W is [R][IN][OUT])
__global__ void wconv_kernel(const float* __restrict__ W, bf16* __restrict__ Wt,
                             int IN, int OUT, int total) {
    int i = blockIdx.x * blockDim.x + threadIdx.x;
    if (i >= total) return;
    int per = IN * OUT;
    int r = i / per;
    int rem = i - r * per;
    int k = rem / OUT;
    int o = rem - k * OUT;
    Wt[(size_t)r * per + (size_t)o * IN + k] = (bf16)W[i];
}

// ---------------------------------------------------------------------------
// Edge scatter (float4-granular):
//   Y[dst[e]*W + f..f+3] += X[src[e]*W + f..f+3] * sSrc(src) * sDst(dst)
// rsSrc / rsDst may be null (treated as 1.0).  width = 1<<shift, shift >= 2.
// total = E << (shift-2) threads.
// ---------------------------------------------------------------------------
__global__ void scatter_kernel(const int* __restrict__ src, const int* __restrict__ dst,
                               const float* __restrict__ rsSrc, const float* __restrict__ rsDst,
                               const float* __restrict__ X, float* __restrict__ Y,
                               int shift, int total) {
    int i = blockIdx.x * blockDim.x + threadIdx.x;
    if (i >= total) return;
    const int cshift = shift - 2;                 // float4 chunks per row
    int e  = i >> cshift;
    int f4 = (i & ((1 << cshift) - 1)) << 2;
    int se = src[e];
    int de = dst[e];
    float s = 1.0f;
    if (rsSrc) s *= rsSrc[se];
    if (rsDst) s *= rsDst[de];
    const float4 v = *(const float4*)(X + ((size_t)se << shift) + f4);
    float* y = Y + ((size_t)de << shift) + f4;
    atomicAdd(y + 0, v.x * s);
    atomicAdd(y + 1, v.y * s);
    atomicAdd(y + 2, v.z * s);
    atomicAdd(y + 3, v.w * s);
}

// ---------------------------------------------------------------------------
// WMMA GEMM:  C[N,OUT] = sum_{r<NREL} (rowscale_r ⊙ A_r[N,IN]) @ Wt_r  (+ Σ_r bias_r)
// A_r = A + r*aRelStride (floats), rowscale_r = rscale + r*rsRelStride.
// Wt: bf16, transposed [NREL][OUT][IN].
// Each wave computes NT adjacent 16x16 output tiles (one converted A fragment
// feeds NT WMMAs -> 4x less A traffic), 8 waves per 256-thread block.
// ---------------------------------------------------------------------------
template <int IN, int OUT, int NREL, bool HAS_BIAS, int NT>
__global__ void gemm_wmma(const float* __restrict__ A, size_t aRelStride,
                          const float* __restrict__ rscale, size_t rsRelStride,
                          const bf16* __restrict__ Wt,
                          const float* __restrict__ bias,
                          float* __restrict__ C) {
    const int lane = threadIdx.x & 31;
    const int wave = threadIdx.x >> 5;
    const int tilesPerRow  = OUT / 16;
    const int groupsPerRow = tilesPerRow / NT;
    const int nGroups = (GN_NODES / 16) * groupsPerRow;
    int grp = blockIdx.x * (blockDim.x >> 5) + wave;
    if (grp >= nGroups) return;                 // wave-uniform: EXEC all-ones below
    const int trow = grp / groupsPerRow;
    const int tg   = grp - trow * groupsPerRow;
    const int col0 = tg * NT * 16;              // first output column of this wave

    const int hl   = lane >> 4;                 // half-wave selector
    const int row  = trow * 16 + (lane & 15);   // A row (M) this lane feeds
    const int klo  = hl * 8;                    // A K sub-offset
    const int ln15 = lane & 15;
    const int bk   = hl * 8;                    // B K sub-offset

    v8f acc[NT];
#pragma unroll
    for (int t = 0; t < NT; ++t) acc[t] = (v8f){0.f,0.f,0.f,0.f,0.f,0.f,0.f,0.f};

#pragma unroll
    for (int r = 0; r < NREL; ++r) {
        const float* Ar = A + (size_t)r * aRelStride + (size_t)row * IN;
        const float  s  = rscale[(size_t)r * rsRelStride + row];
        const bf16*  Wr = Wt + (size_t)r * (size_t)OUT * IN;
#pragma unroll
        for (int k0 = 0; k0 < IN; k0 += 32) {
            // ---- A fragment (16-bit A 16x32 layout, ISA §7.12.2) ----
            const float4 a0 = *(const float4*)(Ar + k0 + klo + 0);
            const float4 a1 = *(const float4*)(Ar + k0 + klo + 4);
            const float4 a2 = *(const float4*)(Ar + k0 + klo + 16);
            const float4 a3 = *(const float4*)(Ar + k0 + klo + 20);
            v16bf a;
            a[0]  = (bf16)(a0.x * s); a[1]  = (bf16)(a0.y * s);
            a[2]  = (bf16)(a0.z * s); a[3]  = (bf16)(a0.w * s);
            a[4]  = (bf16)(a1.x * s); a[5]  = (bf16)(a1.y * s);
            a[6]  = (bf16)(a1.z * s); a[7]  = (bf16)(a1.w * s);
            a[8]  = (bf16)(a2.x * s); a[9]  = (bf16)(a2.y * s);
            a[10] = (bf16)(a2.z * s); a[11] = (bf16)(a2.w * s);
            a[12] = (bf16)(a3.x * s); a[13] = (bf16)(a3.y * s);
            a[14] = (bf16)(a3.z * s); a[15] = (bf16)(a3.w * s);
            // ---- NT column tiles share this A fragment ----
#pragma unroll
            for (int t = 0; t < NT; ++t) {
                const int bcol = col0 + t * 16 + ln15;   // B column (N) this lane feeds
                union { uint4 q[2]; v16bf v; } bu;
                const uint4* bq = (const uint4*)(Wr + (size_t)bcol * IN + k0 + bk);
                bu.q[0] = bq[0];
                bu.q[1] = bq[2];
                acc[t] = __builtin_amdgcn_wmma_f32_16x16x32_bf16(
                    false, a, false, bu.v, (short)0, acc[t], false, false);
            }
        }
    }

    // ---- store D (32-bit C/D 16x16 layout): M = 8*hl + v, N = lane&15 ----
    const int cm0 = trow * 16 + hl * 8;
#pragma unroll
    for (int t = 0; t < NT; ++t) {
        const int cn = col0 + t * 16 + ln15;
        float bs = 0.f;
        if (HAS_BIAS) {
#pragma unroll
            for (int r = 0; r < NREL; ++r) bs += bias[r * OUT + cn];
        }
#pragma unroll
        for (int v = 0; v < 8; ++v) {
            C[(size_t)(cm0 + v) * OUT + cn] = acc[t][v] + bs;
        }
    }
}

// ---------------------------------------------------------------------------
// Head: out[o] = sum_j ( mean_f(X3[o*64+j, f]) * Wd[j] ) + bd
// one block of 64 threads per output element o (1600 blocks)
// ---------------------------------------------------------------------------
__global__ void head_kernel(const float* __restrict__ X3, const float* __restrict__ Wd,
                            const float* __restrict__ bd, float* __restrict__ out) {
    __shared__ float red[64];
    const int j = threadIdx.x;
    const size_t n = (size_t)blockIdx.x * 64 + j;
    const float* row = X3 + n * 64;
    float m = 0.f;
#pragma unroll
    for (int f = 0; f < 64; f += 4) {
        float4 v = *(const float4*)(row + f);
        m += v.x + v.y + v.z + v.w;
    }
    red[j] = (m * (1.0f / 64.0f)) * Wd[j];
    __syncthreads();
#pragma unroll
    for (int off = 32; off > 0; off >>= 1) {
        if (j < off) red[j] += red[j + off];
        __syncthreads();
    }
    if (j == 0) out[blockIdx.x] = red[0] + bd[0];
}

// ---------------------------------------------------------------------------
// Host-side orchestration
// ---------------------------------------------------------------------------
static inline int blocks_for(long long n, int t) { return (int)((n + t - 1) / t); }

extern "C" void kernel_launch(void* const* d_in, const int* in_sizes, int n_in,
                              void* d_out, int out_size, void* d_ws, size_t ws_size,
                              hipStream_t stream) {
    (void)in_sizes; (void)n_in; (void)out_size; (void)ws_size;
    const float* h   = (const float*)d_in[0];
    const int*   src = (const int*)  d_in[1];
    const int*   dst = (const int*)  d_in[2];
    const float* W1  = (const float*)d_in[3];
    const float* b1  = (const float*)d_in[4];
    const float* W2  = (const float*)d_in[5];
    const float* b2  = (const float*)d_in[6];
    const float* W3  = (const float*)d_in[7];
    const float* b3  = (const float*)d_in[8];
    const float* Wd  = (const float*)d_in[9];
    const float* bd  = (const float*)d_in[10];
    float* out = (float*)d_out;

    const int N = GN_NODES, R = GN_R, E = GN_E;

    // ---- workspace layout (256B aligned regions) ----
    char* base = (char*)d_ws;
    size_t o = 0;
    auto alloc = [&](size_t bytes) { size_t c = o; o = (o + bytes + 255) & ~(size_t)255; return c; };
    size_t rsqOut_o = alloc((size_t)R * N * 4);            // rsqrt(out-degree) per relation
    size_t rsqIn_o  = alloc((size_t)R * N * 4);            // rsqrt(in-degree)  per relation
    size_t wt1_o    = alloc((size_t)R * 64 * 256 * 2);     // bf16 transposed weights
    size_t wt2_o    = alloc((size_t)R * 256 * 128 * 2);
    size_t wt3_o    = alloc((size_t)R * 128 * 64 * 2);
    size_t ag_o     = alloc((size_t)R * N * 64 * 4);       // L1 agg; reused as yb (L2/L3)
    size_t x1_o     = alloc((size_t)N * 256 * 4);          // L1 out; region reused for x3
    size_t x2_o     = alloc((size_t)N * 128 * 4);          // L2 out
    size_t x3_o     = x1_o;                                 // x1 dead after L2 GEMMs

    float* rsqOut = (float*)(base + rsqOut_o);
    float* rsqIn  = (float*)(base + rsqIn_o);
    bf16*  Wt1    = (bf16*) (base + wt1_o);
    bf16*  Wt2    = (bf16*) (base + wt2_o);
    bf16*  Wt3    = (bf16*) (base + wt3_o);
    float* AG     = (float*)(base + ag_o);
    float* X1     = (float*)(base + x1_o);
    float* X2     = (float*)(base + x2_o);
    float* X3     = (float*)(base + x3_o);
    float* YB     = AG;   // per-relation GEMM output buffer for layers 2/3

    const int T = 256;

    // ---- 0) degrees -> rsqrt(clip(deg,1)) ----
    fill_f32<<<blocks_for(2LL * R * N, T), T, 0, stream>>>(rsqOut, 0.f, 2 * R * N);
    degree_kernel<<<blocks_for((long long)R * E, T), T, 0, stream>>>(src, dst, rsqOut, rsqIn);
    rsqrt_clip_kernel<<<blocks_for(2LL * R * N, T), T, 0, stream>>>(rsqOut, 2 * R * N);

    // ---- weights -> bf16 transposed ----
    wconv_kernel<<<blocks_for((long long)R * 64 * 256, T), T, 0, stream>>>(W1, Wt1, 64, 256, R * 64 * 256);
    wconv_kernel<<<blocks_for((long long)R * 256 * 128, T), T, 0, stream>>>(W2, Wt2, 256, 128, R * 256 * 128);
    wconv_kernel<<<blocks_for((long long)R * 128 * 64, T), T, 0, stream>>>(W3, Wt3, 128, 64, R * 128 * 64);

    // ---- Layer 1: aggregate-first (width 64), then fused 3-relation GEMM 64->256 ----
    fill_f32<<<blocks_for((long long)R * N * 64, T), T, 0, stream>>>(AG, 0.f, R * N * 64);
    for (int r = 0; r < R; ++r) {
        scatter_kernel<<<blocks_for((long long)E * 16, T), T, 0, stream>>>(
            src + (size_t)r * E, dst + (size_t)r * E,
            rsqOut + (size_t)r * N, nullptr,
            h, AG + (size_t)r * N * 64, 6, E * 16);
    }
    {
        int groups = (N / 16) * (256 / 16 / 4);
        gemm_wmma<64, 256, GN_R, true, 4><<<groups / 8, T, 0, stream>>>(
            AG, (size_t)N * 64, rsqIn, (size_t)N, Wt1, b1, X1);
    }

    // ---- Layer 2: weight-first 256->128, per-relation GEMM + normalized scatter ----
    bias_init_kernel<<<blocks_for((long long)N * 128, T), T, 0, stream>>>(X2, b2, 128, N * 128);
    for (int r = 0; r < R; ++r) {
        int groups = (N / 16) * (128 / 16 / 4);
        gemm_wmma<256, 128, 1, false, 4><<<groups / 8, T, 0, stream>>>(
            X1, 0, rsqOut + (size_t)r * N, 0, Wt2 + (size_t)r * 256 * 128, nullptr, YB);
        scatter_kernel<<<blocks_for((long long)E * 32, T), T, 0, stream>>>(
            src + (size_t)r * E, dst + (size_t)r * E,
            nullptr, rsqIn + (size_t)r * N,
            YB, X2, 7, E * 32);
    }

    // ---- Layer 3: weight-first 128->64 ----
    bias_init_kernel<<<blocks_for((long long)N * 64, T), T, 0, stream>>>(X3, b3, 64, N * 64);
    for (int r = 0; r < R; ++r) {
        int groups = (N / 16) * (64 / 16 / 4);
        gemm_wmma<128, 64, 1, false, 4><<<groups / 8, T, 0, stream>>>(
            X2, 0, rsqOut + (size_t)r * N, 0, Wt3 + (size_t)r * 128 * 64, nullptr, YB);
        scatter_kernel<<<blocks_for((long long)E * 16, T), T, 0, stream>>>(
            src + (size_t)r * E, dst + (size_t)r * E,
            nullptr, rsqIn + (size_t)r * N,
            YB, X3, 6, E * 16);
    }

    // ---- Head: mean over 64 features + Linear(64,1) per 64-node group ----
    head_kernel<<<N / 64, 64, 0, stream>>>(X3, Wd, bd, out);
}
